// LLaMALayer_71064528880297
// MI455X (gfx1250) — compile-verified, matrix operand order checked by reading, not run on hardware
//
#include <hip/hip_runtime.h>
#include <math.h>

// ---------------------------------------------------------------------------
// LLaMA layer for MI455X (gfx1250). ~0.9 TFLOP, ~800MB weight traffic ->
// compute bound: all GEMMs + attention run on v_wmma_f32_16x16x32_bf16.
// GEMM is double-buffered and uses gfx1250 async global->LDS copies
// (ASYNCcnt) for the bf16 A tiles; fp32 weights are register-prefetched and
// converted to bf16 while the WMMA pipe drains the previous tile.
// ---------------------------------------------------------------------------

enum { kS = 2048, kD = 4096, kH = 32, kDH = 128, kDFF = 10922, kDFFP = 10944 };

typedef __attribute__((ext_vector_type(16))) __bf16 v16bf;
typedef __attribute__((ext_vector_type(8)))  float  v8f;
typedef __attribute__((ext_vector_type(4)))  int    v4i;

#if defined(__has_builtin)
#if __has_builtin(__builtin_amdgcn_global_load_async_to_lds_b128)
#define HAVE_ASYNC_LDS 1
#else
#define HAVE_ASYNC_LDS 0
#endif
#else
#define HAVE_ASYNC_LDS 0
#endif

__device__ __forceinline__ unsigned short f32_to_bf16_rte(float f) {
  union { float f; unsigned int u; } a; a.f = f;
  unsigned int r = a.u + 0x7FFFu + ((a.u >> 16) & 1u);
  return (unsigned short)(r >> 16);
}

// 16-byte global->LDS copy; async (ASYNCcnt-tracked, no VGPR staging) when the
// toolchain exposes the gfx1250 builtin. Signature (from hipcc diagnostics):
// (v4i addrspace(1)*, v4i addrspace(3)*, imm offset, imm cpol).
__device__ __forceinline__ void copy16_g2l(const unsigned short* g, unsigned short* l) {
#if HAVE_ASYNC_LDS
  __builtin_amdgcn_global_load_async_to_lds_b128(
      (__attribute__((address_space(1))) v4i*)g,
      (__attribute__((address_space(3))) v4i*)l, 0, 0);
#else
  *reinterpret_cast<float4*>(l) = *reinterpret_cast<const float4*>(g);
#endif
}

__device__ __forceinline__ void wait_async_copies() {
#if HAVE_ASYNC_LDS
#if __has_builtin(__builtin_amdgcn_s_wait_asynccnt)
  __builtin_amdgcn_s_wait_asynccnt(0);
#else
  asm volatile("s_wait_asynccnt 0" ::: "memory");
#endif
#endif
}

// A/B fragment for V_WMMA_*_16X16X32 (16-bit): per ISA 7.12.2, lane (0..31)
// holds M(or N)=lane&15; elements e=0..7 -> K = laneHi*8 + e,
// e=8..15 -> K = 16 + laneHi*8 + (e-8). Two contiguous 16B chunks per lane.
__device__ __forceinline__ v16bf load_frag_b16(const unsigned short* rowptr, int laneHi) {
  union { v16bf v; float4 f[2]; } u;
  u.f[0] = *reinterpret_cast<const float4*>(rowptr + laneHi * 8);
  u.f[1] = *reinterpret_cast<const float4*>(rowptr + 16 + laneHi * 8);
  return u.v;
}

__device__ __forceinline__ v8f wmma_bf16(v16bf a, v16bf b, v8f c) {
  return __builtin_amdgcn_wmma_f32_16x16x32_bf16(false, a, false, b, (short)0, c, false, false);
}

// ---------------------------------------------------------------------------
// GEMM: C[M,N] = A[M,K](bf16) * B[K,N](fp32 weights, cast to bf16 in LDS)
// Block tile 128x128, K-step 32, 256 threads = 8 waves, each wave 2x4 tiles.
// Double buffered: A via async global->LDS, B via register prefetch + convert.
// MODE 0: Cf = acc + bias
// MODE 1: Cf = aux + acc + bias            (residual add)
// MODE 2: Cb = bf16(aux * silu(acc+bias))  (SwiGLU fuse; aux = h = xn@w1+b1)
// ---------------------------------------------------------------------------
template<int MODE>
__global__ __launch_bounds__(256) void gemm_bf16_wmma(
    const unsigned short* __restrict__ A,
    const float* __restrict__ B,
    const float* __restrict__ bias,
    const float* __restrict__ aux,
    float* __restrict__ Cf,
    unsigned short* __restrict__ Cb,
    int M, int N, int K, int Kb, int ldc)
{
  __shared__ __align__(16) unsigned short As[2][128 * 32];   // [m][k]
  __shared__ __align__(16) unsigned short Bs[2][128 * 32];   // [n][k] (transposed)
  const int tid    = threadIdx.x;
  const int wave   = tid >> 5;
  const int lane   = tid & 31;
  const int laneHi = lane >> 4;
  const int lane16 = lane & 15;
  const int m0 = blockIdx.y * 128;
  const int n0 = blockIdx.x * 128;
  const int wm = (wave & 3) * 32;    // 4 waves across M (32 rows each)
  const int wn = (wave >> 2) * 64;   // 2 waves across N (64 cols each)

  v8f acc[2][4];
  for (int i = 0; i < 2; ++i)
    for (int j = 0; j < 4; ++j)
      for (int e = 0; e < 8; ++e) acc[i][j][e] = 0.0f;

  const int ar = tid >> 1;          // A: 2 threads per row, 16 halfs each
  const int ac = (tid & 1) * 16;
  const int bk = tid >> 3;          // B: 8 threads per k-row, 16 cols each
  const int bc = (tid & 7) * 16;
  const unsigned short* Abase = A + (size_t)(m0 + ar) * K + ac;

  float bregs[16];

  auto load_a_tile = [&](int k0, int buf) {
    copy16_g2l(Abase + k0, &As[buf][ar * 32 + ac]);
  };
  auto load_b_regs = [&](int k0) {
    const int gk = k0 + bk;
    const float* Bg = B + (size_t)gk * N + (n0 + bc);
    #pragma unroll
    for (int i = 0; i < 16; ++i) {
      const int gn = n0 + bc + i;
      bregs[i] = (gk < Kb && gn < N) ? Bg[i] : 0.0f;
    }
    // stream hint for the tile after next (weights are read exactly once)
    if (k0 + 32 + bk < Kb)
      __builtin_prefetch(B + (size_t)(k0 + 32 + bk) * N + (n0 + bc), 0, 0);
  };
  auto store_b_tile = [&](int buf) {
    #pragma unroll
    for (int i = 0; i < 16; ++i)
      Bs[buf][(bc + i) * 32 + bk] = f32_to_bf16_rte(bregs[i]);
  };

  // prologue: stage first K-step
  load_a_tile(0, 0);
  load_b_regs(0);
  store_b_tile(0);
  wait_async_copies();
  __syncthreads();

  int buf = 0;
  for (int k0 = 0; k0 < K; k0 += 32) {
    const bool has_next = (k0 + 32 < K);
    if (has_next) {
      load_a_tile(k0 + 32, buf ^ 1);   // async copy overlaps the WMMAs below
      load_b_regs(k0 + 32);            // fp32 weight prefetch into registers
    }

    v16bf afrag[2], bfrag[4];
    #pragma unroll
    for (int i = 0; i < 2; ++i)
      afrag[i] = load_frag_b16(&As[buf][(wm + i * 16 + lane16) * 32], laneHi);
    #pragma unroll
    for (int j = 0; j < 4; ++j)
      bfrag[j] = load_frag_b16(&Bs[buf][(wn + j * 16 + lane16) * 32], laneHi);
    #pragma unroll
    for (int i = 0; i < 2; ++i)
      #pragma unroll
      for (int j = 0; j < 4; ++j)
        acc[i][j] = wmma_bf16(afrag[i], bfrag[j], acc[i][j]);

    if (has_next) store_b_tile(buf ^ 1);  // convert while WMMAs drain
    wait_async_copies();
    __syncthreads();
    buf ^= 1;
  }

  // Epilogue. C/D layout: VGPR r -> M = r + laneHi*8, N = lane&15 (ISA 7.12.2).
  #pragma unroll
  for (int i = 0; i < 2; ++i) {
    #pragma unroll
    for (int j = 0; j < 4; ++j) {
      const int n = n0 + wn + j * 16 + lane16;
      if (n >= N) continue;
      const float bn = bias[n];
      #pragma unroll
      for (int r = 0; r < 8; ++r) {
        const int m = m0 + wm + i * 16 + r + laneHi * 8;
        float v = acc[i][j][r] + bn;
        const size_t aidx = (size_t)m * N + n;
        if (MODE == 0) {
          Cf[(size_t)m * ldc + n] = v;
        } else if (MODE == 1) {
          Cf[(size_t)m * ldc + n] = aux[aidx] + v;
        } else {
          const float s = v / (1.0f + __expf(-v));       // silu(g)
          Cb[(size_t)m * ldc + n] = f32_to_bf16_rte(aux[aidx] * s);
        }
      }
    }
  }
}

// ---------------------------------------------------------------------------
// Flash-style causal attention, one (head, 64-query block) per 256-thread block.
// scores = Q.K^T via WMMA; online softmax in LDS; O += P.V via WMMA.
// ---------------------------------------------------------------------------
__global__ __launch_bounds__(256) void attn_wmma(
    const unsigned short* __restrict__ Qg,
    const unsigned short* __restrict__ Kg,
    const unsigned short* __restrict__ Vg,
    unsigned short* __restrict__ Og)
{
  __shared__ __align__(16) unsigned short Qs[64 * 128];   // [q][dh]   (A)
  __shared__ __align__(16) unsigned short Ks[64 * 128];   // [key][dh] (B = K^T)
  __shared__ __align__(16) unsigned short Vs[128 * 64];   // [dh][key] (B for P.V)
  __shared__ __align__(16) unsigned short Ps[64 * 64];    // [q][key]  (A for P.V)
  __shared__ float Ss[64 * 65];
  __shared__ float mrow[64], lrow[64], crow[64];

  const int tid    = threadIdx.x;
  const int wave   = tid >> 5;
  const int lane   = tid & 31;
  const int laneHi = lane >> 4;
  const int lane16 = lane & 15;
  const int q0   = blockIdx.x * 64;
  const int head = blockIdx.y;
  const size_t hoff = (size_t)head * kDH;

  {  // load Q tile (bf16, coalesced float4)
    const int row = tid >> 2;
    const int c0  = (tid & 3) * 32;
    const float4* src = reinterpret_cast<const float4*>(Qg + (size_t)(q0 + row) * kD + hoff + c0);
    float4* dst = reinterpret_cast<float4*>(&Qs[row * 128 + c0]);
    #pragma unroll
    for (int i = 0; i < 4; ++i) dst[i] = src[i];
  }
  if (tid < 64) { mrow[tid] = -__builtin_inff(); lrow[tid] = 0.0f; }

  const int mt = wave & 3;    // 16-row M tile owned by this wave
  const int ng = wave >> 2;   // N half (scores: 2 tiles, O: 4 tiles)
  v8f oacc[4];
  for (int j = 0; j < 4; ++j)
    for (int e = 0; e < 8; ++e) oacc[j][e] = 0.0f;

  const int kbmax = q0 / 64;
  for (int kb = 0; kb <= kbmax; ++kb) {
    __syncthreads();
    {  // load K tile row-major, V tile transposed
      const int row = tid >> 2;
      const int c0  = (tid & 3) * 32;
      const float4* src = reinterpret_cast<const float4*>(Kg + (size_t)(kb * 64 + row) * kD + hoff + c0);
      float4* dst = reinterpret_cast<float4*>(&Ks[row * 128 + c0]);
      #pragma unroll
      for (int i = 0; i < 4; ++i) dst[i] = src[i];
      const unsigned short* vsrc = Vg + (size_t)(kb * 64 + row) * kD + hoff + c0;
      #pragma unroll
      for (int i = 0; i < 32; ++i) Vs[(c0 + i) * 64 + row] = vsrc[i];
    }
    __syncthreads();

    {  // scores: each wave computes a 16x32 slab of the 64x64 tile
      v8f sacc[2];
      for (int t = 0; t < 2; ++t)
        for (int e = 0; e < 8; ++e) sacc[t][e] = 0.0f;
      #pragma unroll
      for (int kk = 0; kk < 128; kk += 32) {
        v16bf aq = load_frag_b16(&Qs[(mt * 16 + lane16) * 128 + kk], laneHi);
        #pragma unroll
        for (int t = 0; t < 2; ++t) {
          const int nt = ng * 2 + t;
          v16bf bk_ = load_frag_b16(&Ks[(nt * 16 + lane16) * 128 + kk], laneHi);
          sacc[t] = wmma_bf16(aq, bk_, sacc[t]);
        }
      }
      const float scale = 0.08838834764831845f;  // 1/sqrt(128)
      #pragma unroll
      for (int t = 0; t < 2; ++t) {
        const int nt = ng * 2 + t;
        #pragma unroll
        for (int r = 0; r < 8; ++r)
          Ss[(mt * 16 + r + laneHi * 8) * 65 + nt * 16 + lane16] = sacc[t][r] * scale;
      }
    }
    __syncthreads();

    if (tid < 64) {  // online softmax, one thread per query row
      const int r = tid;
      const int q = q0 + r;
      const int kstart = kb * 64;
      float mx = -__builtin_inff();
      for (int c = 0; c < 64; ++c) {
        float s = Ss[r * 65 + c];
        if (kstart + c > q) s = -__builtin_inff();   // causal mask
        Ss[r * 65 + c] = s;
        mx = fmaxf(mx, s);
      }
      const float mold = mrow[r];
      const float mnew = fmaxf(mold, mx);
      float corr = 1.0f, sum = 0.0f;
      if (mnew == -__builtin_inff()) {
        for (int c = 0; c < 64; ++c) Ps[r * 64 + c] = 0;
      } else {
        corr = __expf(mold - mnew);                  // exp(-inf)=0 on first hit
        for (int c = 0; c < 64; ++c) {
          const float s = Ss[r * 65 + c];
          const float p = (s == -__builtin_inff()) ? 0.0f : __expf(s - mnew);
          sum += p;
          Ps[r * 64 + c] = f32_to_bf16_rte(p);
        }
      }
      mrow[r] = mnew;
      lrow[r] = lrow[r] * corr + sum;
      crow[r] = corr;
    }
    __syncthreads();

    {  // rescale O accumulators, then O += P.V
      float f[8];
      #pragma unroll
      for (int r = 0; r < 8; ++r) f[r] = crow[mt * 16 + r + laneHi * 8];
      #pragma unroll
      for (int j = 0; j < 4; ++j)
        #pragma unroll
        for (int r = 0; r < 8; ++r) oacc[j][r] *= f[r];
      #pragma unroll
      for (int kk = 0; kk < 64; kk += 32) {
        v16bf ap = load_frag_b16(&Ps[(mt * 16 + lane16) * 64 + kk], laneHi);
        #pragma unroll
        for (int j = 0; j < 4; ++j) {
          const int nt = ng * 4 + j;
          v16bf bv_ = load_frag_b16(&Vs[(nt * 16 + lane16) * 64 + kk], laneHi);
          oacc[j] = wmma_bf16(ap, bv_, oacc[j]);
        }
      }
    }
  }
  __syncthreads();

  #pragma unroll
  for (int j = 0; j < 4; ++j) {
    const int n = (ng * 4 + j) * 16 + lane16;
    #pragma unroll
    for (int r = 0; r < 8; ++r) {
      const int m = mt * 16 + r + laneHi * 8;
      const float inv = 1.0f / lrow[m];
      Og[(size_t)(q0 + m) * kD + hoff + n] = f32_to_bf16_rte(oacc[j][r] * inv);
    }
  }
}

// ---------------------------------------------------------------------------
__global__ __launch_bounds__(256) void rmsnorm_bf16(
    const float* __restrict__ x, const float* __restrict__ scale,
    unsigned short* __restrict__ out)
{
  __shared__ float red[256];
  const int row = blockIdx.x;
  const int tid = threadIdx.x;
  const float* xr = x + (size_t)row * kD;
  float s = 0.0f;
  for (int c = tid; c < kD; c += 256) { const float v = xr[c]; s += v * v; }
  red[tid] = s;
  __syncthreads();
  for (int st = 128; st > 0; st >>= 1) {
    if (tid < st) red[tid] += red[tid + st];
    __syncthreads();
  }
  const float inv = rsqrtf(red[0] / (float)kD + 1e-6f);
  unsigned short* orow = out + (size_t)row * kD;
  for (int c = tid; c < kD; c += 256)
    orow[c] = f32_to_bf16_rte(scale[c] * xr[c] * inv);
}

// RoPE on q,k (biases already applied in the GEMM epilogue) + bf16 cast.
__global__ __launch_bounds__(256) void rope_cast(
    const float* __restrict__ qf, const float* __restrict__ kf, const float* __restrict__ vf,
    unsigned short* __restrict__ Qb, unsigned short* __restrict__ Kb, unsigned short* __restrict__ Vb)
{
  const int idx = blockIdx.x * 256 + threadIdx.x;
  if (idx >= kS * (kD / 2)) return;
  const int s = idx >> 11;        // kD/2 = 2048 pairs per row
  const int p = idx & 2047;
  const int h = p >> 6;           // 64 pairs per head
  const int i = p & 63;
  const size_t base = (size_t)s * kD + h * kDH + 2 * i;
  // theta_i = 10000^-(2i/128)
  const float ang = (float)s * __expf(-((float)(2 * i) / (float)kDH) * 9.210340371976184f);
  float sn, cs;
  __sincosf(ang, &sn, &cs);
  {
    const float e = qf[base], o = qf[base + 1];
    Qb[base]     = f32_to_bf16_rte(e * cs - o * sn);
    Qb[base + 1] = f32_to_bf16_rte(e * sn + o * cs);
  }
  {
    const float e = kf[base], o = kf[base + 1];
    Kb[base]     = f32_to_bf16_rte(e * cs - o * sn);
    Kb[base + 1] = f32_to_bf16_rte(e * sn + o * cs);
  }
  Vb[base]     = f32_to_bf16_rte(vf[base]);
  Vb[base + 1] = f32_to_bf16_rte(vf[base + 1]);
}

// zero the K-padding columns of the SwiGLU activation (DFF..DFFP)
__global__ void zero_pad_ffb(unsigned short* __restrict__ ffb) {
  const int padw = kDFFP - kDFF;
  const int idx = blockIdx.x * 256 + threadIdx.x;
  if (idx >= kS * padw) return;
  const int r = idx / padw;
  const int c = kDFF + idx % padw;
  ffb[(size_t)r * kDFFP + c] = 0;
}

// ---------------------------------------------------------------------------
extern "C" void kernel_launch(void* const* d_in, const int* in_sizes, int n_in,
                              void* d_out, int out_size, void* d_ws, size_t ws_size,
                              hipStream_t stream) {
  (void)in_sizes; (void)n_in; (void)out_size; (void)ws_size;
  const float* x      = (const float*)d_in[0];
  const float* wq     = (const float*)d_in[1];
  const float* bq     = (const float*)d_in[2];
  const float* wk     = (const float*)d_in[3];
  const float* bk     = (const float*)d_in[4];
  const float* wv     = (const float*)d_in[5];
  const float* bv     = (const float*)d_in[6];
  const float* wo     = (const float*)d_in[7];
  const float* bo     = (const float*)d_in[8];
  const float* scale1 = (const float*)d_in[9];
  const float* scale2 = (const float*)d_in[10];
  const float* w1     = (const float*)d_in[11];
  const float* b1     = (const float*)d_in[12];
  const float* wg     = (const float*)d_in[13];
  const float* bg     = (const float*)d_in[14];
  const float* w2     = (const float*)d_in[15];
  const float* b2     = (const float*)d_in[16];
  float* out = (float*)d_out;

  char* ws = (char*)d_ws;
  size_t off = 0;
  auto alloc = [&](size_t bytes) -> void* {
    void* p = ws + off;
    off += (bytes + 255) & ~(size_t)255;
    return p;
  };
  unsigned short* xn1  = (unsigned short*)alloc((size_t)kS * kD * 2);
  float*          qf   = (float*)alloc((size_t)kS * kD * 4);
  float*          kf   = (float*)alloc((size_t)kS * kD * 4);
  float*          vf   = (float*)alloc((size_t)kS * kD * 4);
  unsigned short* Qb   = (unsigned short*)alloc((size_t)kS * kD * 2);
  unsigned short* Kb   = (unsigned short*)alloc((size_t)kS * kD * 2);
  unsigned short* Vb   = (unsigned short*)alloc((size_t)kS * kD * 2);
  unsigned short* Ob   = (unsigned short*)alloc((size_t)kS * kD * 2);
  float*          x1   = (float*)alloc((size_t)kS * kD * 4);
  unsigned short* xn2  = (unsigned short*)alloc((size_t)kS * kD * 2);
  float*          hbuf = (float*)alloc((size_t)kS * kDFF * 4);
  unsigned short* ffb  = (unsigned short*)alloc((size_t)kS * kDFFP * 2);

  const dim3 blk(256);
  const dim3 gD(kD / 128, kS / 128);                 // N=4096 GEMMs
  const dim3 gF((kDFF + 127) / 128, kS / 128);       // N=10922 GEMMs

  // 1) xn1 = rmsnorm(x, scale1)
  rmsnorm_bf16<<<kS, blk, 0, stream>>>(x, scale1, xn1);
  // 2-4) q,k,v = xn1 @ w{q,k,v} + b
  gemm_bf16_wmma<0><<<gD, blk, 0, stream>>>(xn1, wq, bq, nullptr, qf, nullptr, kS, kD, kD, kD, kD);
  gemm_bf16_wmma<0><<<gD, blk, 0, stream>>>(xn1, wk, bk, nullptr, kf, nullptr, kS, kD, kD, kD, kD);
  gemm_bf16_wmma<0><<<gD, blk, 0, stream>>>(xn1, wv, bv, nullptr, vf, nullptr, kS, kD, kD, kD, kD);
  // 5) RoPE + bf16 cast
  rope_cast<<<(kS * (kD / 2) + 255) / 256, blk, 0, stream>>>(qf, kf, vf, Qb, Kb, Vb);
  // 6) causal attention
  attn_wmma<<<dim3(kS / 64, kH), blk, 0, stream>>>(Qb, Kb, Vb, Ob);
  // 7) x1 = x + Ob @ wo + bo
  gemm_bf16_wmma<1><<<gD, blk, 0, stream>>>(Ob, wo, bo, x, x1, nullptr, kS, kD, kD, kD, kD);
  // 8) xn2 = rmsnorm(x1, scale2)
  rmsnorm_bf16<<<kS, blk, 0, stream>>>(x1, scale2, xn2);
  // 9) h = xn2 @ w1 + b1
  gemm_bf16_wmma<0><<<gF, blk, 0, stream>>>(xn2, w1, b1, nullptr, hbuf, nullptr, kS, kDFF, kD, kD, kDFF);
  // 10) zero ffb K-padding, then ffb = bf16(h * silu(xn2 @ wg + bg))
  zero_pad_ffb<<<(kS * (kDFFP - kDFF) + 255) / 256, blk, 0, stream>>>(ffb);
  gemm_bf16_wmma<2><<<gF, blk, 0, stream>>>(xn2, wg, bg, hbuf, nullptr, ffb, kS, kDFF, kD, kD, kDFFP);
  // 11) out = x1 + ffb @ w2 + b2   (K padded to 10944; w2 staged with k<10922 guard)
  gemm_bf16_wmma<1><<<gD, blk, 0, stream>>>(ffb, w2, b2, x1, out, nullptr, kS, kD, kDFFP, kDFF, kD);
}
